// CTLNetworks_74285754352185
// MI455X (gfx1250) — matrix operand order repaired and checked by reading, not compiled, not run on hardware
//
#include <hip/hip_runtime.h>
#include <hip/hip_bf16.h>
#include <stdint.h>
#include <stddef.h>

typedef _Float16 h16;
typedef __attribute__((ext_vector_type(16))) _Float16 v16h;
typedef __attribute__((ext_vector_type(8)))  float    v8f;

#define BN_SCALE 0.9999950000374997f   /* 1/sqrt(1+1e-5) */
#define MODE_CBR  0   /* relu(bn(conv)) */
#define MODE_CRB  1   /* bn(relu(conv)) */
#define MODE_TANH 2   /* bn(tanh(conv)) */

// ---------------------------------------------------------------------------
// Weight pack: f32 OIHW weights -> WMMA A-fragment layout (contiguous K per lane half, B matches),
// K ordered as (kh,kw,ci) so B-fragment gathers are constant-stride.
// Lane l<16 : V0..7 hold K=0..15 (pairs per dword)
// Lane l>=16: V0..7 hold K=16..31
// ---------------------------------------------------------------------------
__global__ __launch_bounds__(256) void k_pack_w(
    const float* __restrict__ w, h16* __restrict__ apack,
    int Cout, int Cin, int KH, int KW, int KB, int Kgemm, int lc, int total)
{
  int gid = blockIdx.x * blockDim.x + threadIdx.x;
  if (gid >= total) return;
  int lane = gid & 31;
  int blk  = gid >> 5;
  int kb   = blk % KB;
  int mt   = blk / KB;
  int lo = lane & 15, hi = lane >> 4;
  int m  = mt * 16 + lo;
  #pragma unroll
  for (int jj = 0; jj < 8; ++jj) {
    #pragma unroll
    for (int t = 0; t < 2; ++t) {
      int kl = 16 * hi + 2 * jj + t;
      int k  = kb * 32 + kl;
      float val = 0.f;
      if (m < Cout && k < Kgemm) {
        int ci = k & (Cin - 1);
        int s  = k >> lc;            // (kh*KW + kw)
        int kh = s / KW;
        int kw = s - kh * KW;
        val = w[((m * Cin + ci) * KH + kh) * KW + kw];
      }
      apack[(size_t)gid * 16 + jj * 2 + t] = (h16)val;
    }
  }
}

// ---------------------------------------------------------------------------
// Implicit-GEMM conv with WMMA f32_16x16x32_f16.
//   D[Cout x N] = W[Cout x K] * im2col(X)[K x N], N = Nb*H*W, K = Cin*KH*KW
// One wave computes an (MT*16) x 16 output tile: MT A-fragments amortize a
// single (expensive) B gather per k-block.  8 waves / 256-thread block.
// Supports channel concat (X0 = first C0 channels, X1 = rest); C0 and the
// fragment channel base are both multiples of 16 in the fast path, so each
// 16-channel fragment lives entirely in one tensor (hoisted base select).
// SAME padding, stride 1.  All tile guards are wave-uniform so EXEC is all-1s
// at every v_wmma, as the ISA requires.
// ---------------------------------------------------------------------------
template <int MT>
__global__ __launch_bounds__(256) void k_conv_wmma(
    const h16* __restrict__ apack,
    const h16* __restrict__ X0, const h16* __restrict__ X1,
    const float* __restrict__ bias, const float* __restrict__ g,
    const float* __restrict__ be,
    float* __restrict__ Yf, h16* __restrict__ Yh,
    int C0, int Cin, int lc, int H, int W, int KH, int KW,
    int Cout, int KB, int Kgemm, int Ntiles, int Ngemm, int mode)
{
  int wave = threadIdx.x >> 5;
  int lane = threadIdx.x & 31;
  int nt  = blockIdx.x * 8 + wave;     // wave-uniform
  int mt0 = blockIdx.y * MT;
  if (nt >= Ntiles) return;            // uniform per wave -> EXEC stays all-1s

  int lo = lane & 15, hi = lane >> 4;
  int n  = nt * 16 + lo;
  bool nv = (n < Ngemm);
  int HW = H * W;
  int nn = nv ? n : 0;
  int wc = nn % W;
  int h  = (nn / W) % H;
  int b  = nn / HW;
  int khc = KH >> 1, kwc = KW >> 1;
  int KHKW = KH * KW;
  int C1 = Cin - C0;

  v8f acc[MT];
  #pragma unroll
  for (int u = 0; u < MT; ++u) acc[u] = (v8f){};

  for (int kb = 0; kb < KB; ++kb) {
    // A fragments: 32B contiguous per lane -> 2x global_load_b128 each
    v16h a[MT];
    #pragma unroll
    for (int u = 0; u < MT; ++u)
      a[u] = *(const v16h*)(apack + ((size_t)((mt0 + u) * KB + kb) * 32 + lane) * 16);
    if (kb + 1 < KB)  // prefetch next k-block's first A fragment
      __builtin_prefetch(apack + ((size_t)(mt0 * KB + kb + 1) * 32 + lane) * 16, 0, 1);

    v16h bf = {};
    int kbase = kb * 32 + hi * 16;
    if (Cin >= 16) {
      // 16 consecutive K share one (kh,kw): one decode, constant-stride walk
      int s = kbase >> lc;
      if (nv && s < KHKW) {
        int kh = s / KW;
        int kw = s - kh * KW;
        int hin = h + kh - khc;
        int win = wc + kw - kwc;
        if ((unsigned)hin < (unsigned)H && (unsigned)win < (unsigned)W) {
          int ci0  = kbase & (Cin - 1);      // multiple of 16
          int spat = hin * W + win;
          const h16* base = (ci0 < C0)
              ? (X0 + ((b * C0 + ci0) * HW + spat))
              : (X1 + ((b * C1 + (ci0 - C0)) * HW + spat));
          #pragma unroll
          for (int j = 0; j < 16; ++j) bf[j] = base[j * HW];
        }
      }
    } else {
      // tiny-Cin path (cu1: Cin=1)
      #pragma unroll
      for (int j = 0; j < 16; ++j) {
        int k = kbase + j;
        h16 val = (h16)0.f;
        if (nv && k < Kgemm) {
          int ci = k & (Cin - 1);
          int s  = k >> lc;
          int kh = s / KW;
          int kw = s - kh * KW;
          int hin = h + kh - khc;
          int win = wc + kw - kwc;
          if ((unsigned)hin < (unsigned)H && (unsigned)win < (unsigned)W) {
            val = (ci < C0) ? X0[(b * C0 + ci) * HW + hin * W + win]
                            : X1[(b * C1 + (ci - C0)) * HW + hin * W + win];
          }
        }
        bf[j] = val;
      }
    }
    #pragma unroll
    for (int u = 0; u < MT; ++u)
      acc[u] = __builtin_amdgcn_wmma_f32_16x16x32_f16(false, a[u], false, bf,
                                                      (short)0, acc[u], false, false);
  }

  if (!nv) return;
  #pragma unroll
  for (int u = 0; u < MT; ++u) {
    #pragma unroll
    for (int r = 0; r < 8; ++r) {
      int m = (mt0 + u) * 16 + hi * 8 + r;   // C/D layout: lanes 16-31 hold M+8
      if (m < Cout) {
        float v = acc[u][r] + bias[m];
        float gv = g[m], bev = be[m];
        if (mode == MODE_CBR)      v = fmaxf(gv * (v * BN_SCALE) + bev, 0.f);
        else if (mode == MODE_CRB) v = gv * (fmaxf(v, 0.f) * BN_SCALE) + bev;
        else                       v = gv * (tanhf(v) * BN_SCALE) + bev;
        int idx = ((b * Cout + m) * H + h) * W + wc;
        if (Yf) Yf[idx] = v;
        if (Yh) Yh[idx] = (h16)v;
      }
    }
  }
}

// --------------------------- elementwise kernels ---------------------------
__global__ void k_cvt(const float* __restrict__ x, h16* __restrict__ y, int n) {
  int i = blockIdx.x * blockDim.x + threadIdx.x;
  if (i < n) y[i] = (h16)x[i];
}

__global__ void k_scale_cvt(const float* __restrict__ x, h16* __restrict__ y,
                            float s, int n) {
  int i = blockIdx.x * blockDim.x + threadIdx.x;
  if (i < n) y[i] = (h16)(s * x[i]);
}

__global__ void k_up2(const float* __restrict__ src, float* __restrict__ df,
                      h16* __restrict__ dh, int NC, int H2, int W2) {
  int i = blockIdx.x * blockDim.x + threadIdx.x;
  int n = NC * H2 * W2;
  if (i >= n) return;
  int w2 = i % W2;
  int h2 = (i / W2) % H2;
  int nc = i / (W2 * H2);
  float v = src[(nc * (H2 >> 1) + (h2 >> 1)) * (W2 >> 1) + (w2 >> 1)];
  if (df) df[i] = v;
  if (dh) dh[i] = (h16)v;
}

__global__ void k_up2_add_relu(const float* __restrict__ sm,
                               const float* __restrict__ big,
                               float* __restrict__ df, h16* __restrict__ dh,
                               int NC, int H2, int W2) {
  int i = blockIdx.x * blockDim.x + threadIdx.x;
  int n = NC * H2 * W2;
  if (i >= n) return;
  int w2 = i % W2;
  int h2 = (i / W2) % H2;
  int nc = i / (W2 * H2);
  float v = sm[(nc * (H2 >> 1) + (h2 >> 1)) * (W2 >> 1) + (w2 >> 1)] + big[i];
  v = fmaxf(v, 0.f);
  if (df) df[i] = v;
  if (dh) dh[i] = (h16)v;
}

__global__ void k_maxpool2(const float* __restrict__ src, float* __restrict__ df,
                           h16* __restrict__ dh, int NC, int Ho, int Wo) {
  int i = blockIdx.x * blockDim.x + threadIdx.x;
  int n = NC * Ho * Wo;
  if (i >= n) return;
  int wo = i % Wo;
  int ho = (i / Wo) % Ho;
  int nc = i / (Wo * Ho);
  int Wi = Wo * 2;
  const float* p = src + (nc * Ho * 2 + ho * 2) * Wi + wo * 2;
  float v = fmaxf(fmaxf(p[0], p[1]), fmaxf(p[Wi], p[Wi + 1]));
  if (df) df[i] = v;
  if (dh) dh[i] = (h16)v;
}

// Sequential patch scatter (honors overwrite order over t, round-half-even).
__global__ void k_scatter(h16* __restrict__ feat, const float* __restrict__ patches,
                          int strideB, int strideT,
                          const int* __restrict__ locs, const int* __restrict__ tnum) {
  const int T = 6;
  int b = blockIdx.x;
  int tn = tnum[b];
  for (int t = 0; t < T; ++t) {
    if (t < tn) {
      int lx = locs[(b * T + t) * 2 + 0];
      int ly = locs[(b * T + t) * 2 + 1];
      int rx = (lx >> 1) + ((lx & 1) & ((lx >> 1) & 1));  // round-half-even
      int ry = (ly >> 1) + ((ly & 1) & ((ly >> 1) & 1));
      int ix = rx - 16; if (ix < 0) ix = 0; if (rx + 16 > 256) ix = 223;
      int iy = ry - 4;  if (iy < 0) iy = 0; if (ry + 4  > 64)  iy = 55;
      const float* p = patches + b * strideB + t * strideT;
      for (int e = threadIdx.x; e < 16 * 32 * 8; e += blockDim.x) {
        int yy = e & 7;
        int xx = (e >> 3) & 31;
        int c  = e >> 8;
        feat[((b * 16 + c) * 256 + ix + xx) * 64 + iy + yy] = (h16)p[e];
      }
    }
    __syncthreads();
  }
}

__global__ void k_reparam(const float* __restrict__ mu, const float* __restrict__ var,
                          const float* __restrict__ eps, h16* __restrict__ z, int n) {
  int i = blockIdx.x * blockDim.x + threadIdx.x;
  if (i < n) z[i] = (h16)(mu[i] + expf(0.5f * var[i]) * eps[i]);
}

__global__ void k_final(const float* __restrict__ x, float* __restrict__ orelu,
                        float* __restrict__ osig, int n) {
  int i = blockIdx.x * blockDim.x + threadIdx.x;
  if (i < n) {
    float v = x[i];
    orelu[i] = fmaxf(v, 0.f);
    osig[i]  = 1.f / (1.f + expf(-v));
  }
}

// ------------------------------- host side ---------------------------------
static inline int cdiv(int a, int b) { return (a + b - 1) / b; }

static void launch_conv(hipStream_t st, h16* apack,
                        const float* w, const float* b, const float* g,
                        const float* be,
                        const h16* x0, const h16* x1, int C0,
                        int Nb, int H, int W, int Cin, int Cout, int K,
                        int mode, float* yf, h16* yh) {
  int KH = K, KW = K;
  int Kg = Cin * KH * KW;
  int KB = (Kg + 31) / 32;
  int Mt = (Cout + 15) / 16;
  int lc = __builtin_ctz(Cin);
  int total = Mt * KB * 32;
  k_pack_w<<<cdiv(total, 256), 256, 0, st>>>(w, apack, Cout, Cin, KH, KW, KB, Kg, lc, total);
  int Ng = Nb * H * W;
  int Nt = (Ng + 15) / 16;
  int MT = (Mt % 2 == 0) ? 2 : 1;      // stacked M-tiles share one B gather
  dim3 grid(cdiv(Nt, 8), Mt / MT);
  if (MT == 2)
    k_conv_wmma<2><<<grid, 256, 0, st>>>(apack, x0, x1 ? x1 : x0, b, g, be, yf, yh,
                                         C0, Cin, lc, H, W, KH, KW, Cout, KB, Kg,
                                         Nt, Ng, mode);
  else
    k_conv_wmma<1><<<grid, 256, 0, st>>>(apack, x0, x1 ? x1 : x0, b, g, be, yf, yh,
                                         C0, Cin, lc, H, W, KH, KW, Cout, KB, Kg,
                                         Nt, Ng, mode);
}

extern "C" void kernel_launch(void* const* d_in, const int* in_sizes, int n_in,
                              void* d_out, int out_size, void* d_ws, size_t ws_size,
                              hipStream_t stream) {
  (void)in_sizes; (void)n_in; (void)out_size; (void)ws_size;
  const float* feat1 = (const float*)d_in[0];
  const float* feat2 = (const float*)d_in[1];
  const float* feat3 = (const float*)d_in[2];
  const float* feat4 = (const float*)d_in[3];
  const float* current_feats = (const float*)d_in[4];
  const int*   current_locs  = (const int*)d_in[5];
  const int*   target_num    = (const int*)d_in[6];
  const float* desire_stft   = (const float*)d_in[7];
  const float* eps_loc = (const float*)d_in[8];
  const float* eps_des = (const float*)d_in[9];
  const float* eps_cur = (const float*)d_in[10];
  const float* noise_cur = (const float*)d_in[11];
  const float* noise_des = (const float*)d_in[12];
  // params flattened in setup_inputs insertion order: each layer = w,b,g,be.
  // L: 0 pu1, 1 pu2, 2 pu3, 3 cu1, 4 cu2, 5 cu3, 6 el_mu, 7 el_var,
  //    8 e1_mu, 9 e1_var, 10 e2_mu, 11 e2_var,
  //    12..16 dec_true {dloc,dfeat,d1,d2,d3}, 17..21 dec_false {...}
  auto PW  = [&](int L) { return (const float*)d_in[13 + 4 * L + 0]; };
  auto PB  = [&](int L) { return (const float*)d_in[13 + 4 * L + 1]; };
  auto PG  = [&](int L) { return (const float*)d_in[13 + 4 * L + 2]; };
  auto PBE = [&](int L) { return (const float*)d_in[13 + 4 * L + 3]; };

  char* ws = (char*)d_ws;
  size_t off = 0;
  auto alloc = [&](size_t bytes) -> size_t {
    size_t o = off; off = (off + bytes + 255) & ~(size_t)255; return o;
  };
  auto F  = [&](size_t o) { return (float*)(ws + o); };
  auto Hp = [&](size_t o) { return (h16*)(ws + o); };

  // ---- persistent region (~73 MB) ----
  size_t o_ap    = alloc(2u << 20);                 // weight-pack scratch (reused)
  size_t o_ra6h  = alloc((size_t)8388608 * 2);
  size_t o_fcurh = alloc((size_t)4194304 * 2);
  size_t o_fdesh = alloc((size_t)4194304 * 2);
  size_t o_zloch = alloc((size_t)8388608 * 2);
  size_t o_zdesh = alloc((size_t)4194304 * 2);
  size_t o_zcurh = alloc((size_t)4194304 * 2);
  size_t o_ori   = alloc((size_t)1048576 * 4);
  size_t o_mu    = alloc((size_t)8388608 * 4);      // overlaid by decoder arena later
  size_t o_var   = alloc((size_t)8388608 * 4);
  size_t arena0  = off;
  h16* ap = Hp(o_ap);

  auto EW = [&](int n) { return dim3(cdiv(n, 256)); };

  // ================= Phase A: refine pyramid (pu1..pu3) =================
  off = arena0;
  size_t o_f4h  = alloc((size_t)1048576 * 2);
  size_t o_ra1  = alloc((size_t)524288 * 4);
  size_t o_ra2h = alloc((size_t)2097152 * 2);
  size_t o_ra3  = alloc((size_t)1048576 * 4);
  size_t o_ra4h = alloc((size_t)4194304 * 2);
  size_t o_ra5  = alloc((size_t)2097152 * 4);

  k_cvt<<<EW(1048576), 256, 0, stream>>>(feat4, Hp(o_f4h), 1048576);
  launch_conv(stream, ap, PW(0), PB(0), PG(0), PBE(0), Hp(o_f4h), nullptr, 256,
              16, 32, 8, 256, 128, 5, MODE_CBR, F(o_ra1), nullptr);
  k_up2_add_relu<<<EW(2097152), 256, 0, stream>>>(F(o_ra1), feat3, nullptr,
                                                  Hp(o_ra2h), 16 * 128, 64, 16);
  launch_conv(stream, ap, PW(1), PB(1), PG(1), PBE(1), Hp(o_ra2h), nullptr, 128,
              16, 64, 16, 128, 64, 5, MODE_CBR, F(o_ra3), nullptr);
  k_up2_add_relu<<<EW(4194304), 256, 0, stream>>>(F(o_ra3), feat2, nullptr,
                                                  Hp(o_ra4h), 16 * 64, 128, 32);
  launch_conv(stream, ap, PW(2), PB(2), PG(2), PBE(2), Hp(o_ra4h), nullptr, 64,
              16, 128, 32, 64, 32, 5, MODE_CBR, F(o_ra5), nullptr);
  k_up2_add_relu<<<EW(8388608), 256, 0, stream>>>(F(o_ra5), feat1, nullptr,
                                                  Hp(o_ra6h), 16 * 32, 256, 64);

  // ============ Phase B: ConcatUpgrade (cu1..cu3) + scatter ============
  off = arena0;
  size_t o_xh = alloc((size_t)24576 * 2);
  size_t o_A  = alloc((size_t)3145728 * 4);
  size_t o_B  = alloc((size_t)3145728 * 4);
  size_t o_Ch = alloc((size_t)3145728 * 2);
  size_t o_fd = alloc((size_t)65536 * 4);
  size_t o_fc = alloc((size_t)786432 * 4);

  k_scale_cvt<<<EW(4194304), 256, 0, stream>>>(noise_cur, Hp(o_fcurh), 0.001f, 4194304);
  k_scale_cvt<<<EW(4194304), 256, 0, stream>>>(noise_des, Hp(o_fdesh), 0.001f, 4194304);

  auto concat_up = [&](const float* xin, int Nb, float* outbuf) {
    int n0 = Nb * 32 * 8;
    k_cvt<<<EW(n0), 256, 0, stream>>>(xin, Hp(o_xh), n0);
    launch_conv(stream, ap, PW(3), PB(3), PG(3), PBE(3), Hp(o_xh), nullptr, 1,
                Nb, 32, 8, 1, 32, 11, MODE_CBR, F(o_A), nullptr);
    k_up2<<<EW(Nb * 32 * 64 * 16), 256, 0, stream>>>(F(o_A), nullptr, Hp(o_Ch),
                                                     Nb * 32, 64, 16);
    launch_conv(stream, ap, PW(4), PB(4), PG(4), PBE(4), Hp(o_Ch), nullptr, 32,
                Nb, 64, 16, 32, 32, 11, MODE_CBR, F(o_B), nullptr);
    k_maxpool2<<<EW(Nb * 32 * 32 * 8), 256, 0, stream>>>(F(o_B), F(o_A), nullptr,
                                                         Nb * 32, 32, 8);
    k_up2<<<EW(Nb * 32 * 64 * 16), 256, 0, stream>>>(F(o_A), nullptr, Hp(o_Ch),
                                                     Nb * 32, 64, 16);
    launch_conv(stream, ap, PW(5), PB(5), PG(5), PBE(5), Hp(o_Ch), nullptr, 32,
                Nb, 64, 16, 32, 16, 11, MODE_CBR, F(o_B), nullptr);
    k_maxpool2<<<EW(Nb * 16 * 32 * 8), 256, 0, stream>>>(F(o_B), outbuf, nullptr,
                                                         Nb * 16, 32, 8);
  };
  concat_up(desire_stft, 16, F(o_fd));      // feat_d  [16,16,32,8]
  concat_up(current_feats, 96, F(o_fc));    // fc      [16,6,16,32,8]

  k_scatter<<<16, 256, 0, stream>>>(Hp(o_fcurh), F(o_fc), 6 * 4096, 4096,
                                    current_locs, target_num);
  k_scatter<<<16, 256, 0, stream>>>(Hp(o_fdesh), F(o_fd), 4096, 0,
                                    current_locs, target_num);

  // ================= Phase C/D: encoders + reparameterize =================
  launch_conv(stream, ap, PW(6), PB(6), PG(6), PBE(6), Hp(o_ra6h), nullptr, 32,
              16, 256, 64, 32, 32, 5, MODE_TANH, F(o_mu), nullptr);
  launch_conv(stream, ap, PW(7), PB(7), PG(7), PBE(7), Hp(o_ra6h), nullptr, 32,
              16, 256, 64, 32, 32, 5, MODE_TANH, F(o_var), nullptr);
  k_reparam<<<EW(8388608), 256, 0, stream>>>(F(o_mu), F(o_var), eps_loc,
                                             Hp(o_zloch), 8388608);
  launch_conv(stream, ap, PW(8), PB(8), PG(8), PBE(8), Hp(o_fdesh), nullptr, 16,
              16, 256, 64, 16, 16, 5, MODE_TANH, F(o_mu), nullptr);
  launch_conv(stream, ap, PW(9), PB(9), PG(9), PBE(9), Hp(o_fdesh), nullptr, 16,
              16, 256, 64, 16, 16, 5, MODE_TANH, F(o_var), nullptr);
  k_reparam<<<EW(4194304), 256, 0, stream>>>(F(o_mu), F(o_var), eps_des,
                                             Hp(o_zdesh), 4194304);
  launch_conv(stream, ap, PW(10), PB(10), PG(10), PBE(10), Hp(o_fcurh), nullptr, 16,
              16, 256, 64, 16, 16, 5, MODE_TANH, F(o_mu), nullptr);
  launch_conv(stream, ap, PW(11), PB(11), PG(11), PBE(11), Hp(o_fcurh), nullptr, 16,
              16, 256, 64, 16, 16, 5, MODE_TANH, F(o_var), nullptr);
  k_reparam<<<EW(4194304), 256, 0, stream>>>(F(o_mu), F(o_var), eps_cur,
                                             Hp(o_zcurh), 4194304);

  // ================= Phase E: decoders (arena overlays mu/var) =============
  float* out = (float*)d_out;
  auto decoder = [&](const h16* zf1, const h16* zf2, int Lb,
                     float* out_relu, float* out_sig) {
    off = o_mu;  // mu/var + earlier arenas are dead: overlay (~160 MB)
    size_t o_flh = alloc((size_t)8388608 * 2);
    size_t o_ffh = alloc((size_t)8388608 * 2);
    size_t o_f1  = alloc((size_t)8388608 * 4);
    size_t o_f2h = alloc((size_t)33554432 * 2);
    size_t o_f3h = alloc((size_t)16777216 * 2);
    launch_conv(stream, ap, PW(Lb + 0), PB(Lb + 0), PG(Lb + 0), PBE(Lb + 0),
                Hp(o_zloch), nullptr, 32, 16, 256, 64, 32, 32, 5, MODE_CRB,
                nullptr, Hp(o_flh));
    launch_conv(stream, ap, PW(Lb + 1), PB(Lb + 1), PG(Lb + 1), PBE(Lb + 1),
                zf1, zf2, 16, 16, 256, 64, 32, 32, 5, MODE_CRB,
                nullptr, Hp(o_ffh));
    launch_conv(stream, ap, PW(Lb + 2), PB(Lb + 2), PG(Lb + 2), PBE(Lb + 2),
                Hp(o_flh), Hp(o_ffh), 32, 16, 256, 64, 64, 32, 7, MODE_CRB,
                F(o_f1), nullptr);
    k_up2<<<EW(33554432), 256, 0, stream>>>(F(o_f1), nullptr, Hp(o_f2h),
                                            16 * 32, 512, 128);
    launch_conv(stream, ap, PW(Lb + 3), PB(Lb + 3), PG(Lb + 3), PBE(Lb + 3),
                Hp(o_f2h), nullptr, 32, 16, 512, 128, 32, 16, 7, MODE_CRB,
                nullptr, Hp(o_f3h));
    launch_conv(stream, ap, PW(Lb + 4), PB(Lb + 4), PG(Lb + 4), PBE(Lb + 4),
                Hp(o_f3h), nullptr, 16, 16, 512, 128, 16, 1, 7, MODE_CRB,
                F(o_ori), nullptr);
    k_final<<<EW(1048576), 256, 0, stream>>>(F(o_ori), out_relu, out_sig, 1048576);
  };
  decoder(Hp(o_zcurh), Hp(o_zcurh), 12, out + 0, out + 1048576);          // ori
  decoder(Hp(o_zdesh), Hp(o_zcurh), 17, out + 2 * 1048576, out + 3 * 1048576); // des
}
